// Policy_17394617548850
// MI455X (gfx1250) — matrix operand order repaired and checked by reading, not compiled, not run on hardware
//
#include <hip/hip_runtime.h>

// ---------------------------------------------------------------------------
// Types
// ---------------------------------------------------------------------------
typedef __attribute__((ext_vector_type(16))) __bf16        v16bf;
typedef __attribute__((ext_vector_type(8)))  float         v8f;
typedef __attribute__((ext_vector_type(4)))  unsigned int  u32x4;
typedef unsigned short u16;

#define BM 128
#define BN 128
#define BK 32
#define LDS_STRIDE 40                 // 32 bf16 payload + 8 pad -> 80B rows, conflict-free b128 reads
#define TILE_ELems (128 * LDS_STRIDE) // 5120 u16 = 10240 B per tile
#define BUF_BYTES  (2 * TILE_ELems * 2) // A + B tile per buffer = 20480 B

__device__ __forceinline__ u16 f2bf(float f) {
    unsigned int u = __float_as_uint(f);
    u += 0x7FFFu + ((u >> 16) & 1u);          // round-to-nearest-even
    return (u16)(u >> 16);
}
__device__ __forceinline__ float bf2f(u16 h) {
    return __uint_as_float(((unsigned int)h) << 16);
}

union FragU { u32x4 u[2]; v16bf v; };

// CDNA5 async global->LDS copy, 16B per lane, tracked by ASYNCcnt.
__device__ __forceinline__ void async_b128(unsigned lds_off, const u16* gptr) {
    asm volatile("global_load_async_to_lds_b128 %0, %1, off"
                 :: "v"(lds_off), "v"((unsigned long long)gptr)
                 : "memory");
}

// ---------------------------------------------------------------------------
// f32 -> bf16 elementwise (input activations)
// ---------------------------------------------------------------------------
__global__ void __launch_bounds__(256)
to_bf16_kernel(const float* __restrict__ in, u16* __restrict__ out, long long n) {
    long long i = (long long)blockIdx.x * 256 + threadIdx.x;
    if (i < n) out[i] = f2bf(in[i]);
}

// ---------------------------------------------------------------------------
// Batched transpose + f32->bf16: in[z][K][N] -> out[z][N][K]
// All weight tensors here have K, N multiples of 32.
// ---------------------------------------------------------------------------
__global__ void __launch_bounds__(256)
transpose_bf16_kernel(const float* __restrict__ in, u16* __restrict__ out, int K, int N) {
    __shared__ float t[32][33];
    long long z = blockIdx.z;
    in  += z * (long long)K * N;
    out += z * (long long)N * K;
    int n0 = blockIdx.x * 32, k0 = blockIdx.y * 32;
    int tx = threadIdx.x & 31, ty = threadIdx.x >> 5;   // 32 x 8
#pragma unroll
    for (int j = 0; j < 4; ++j)
        t[ty + j * 8][tx] = in[(long long)(k0 + ty + j * 8) * N + (n0 + tx)];
    __syncthreads();
#pragma unroll
    for (int j = 0; j < 4; ++j)
        out[(long long)(n0 + ty + j * 8) * K + (k0 + tx)] = f2bf(t[tx][ty + j * 8]);
}

// ---------------------------------------------------------------------------
// Generic batched bf16 WMMA GEMM, async double-buffered, fused epilogue.
//   C[m, z, n] = epi( A[z](MxK) * Bt[z](NxK)^T )
//   epi: v = (acc + bias[z,n]) * scale; v += extra[m,z,n]; v = elu?(v)
// M multiple of 128; K multiple of 64 (>= 2 stages); N arbitrary (stores
// guarded; OOB Bt rows are clamped to N-1 so every wave issues a uniform
// 4 async loads per stage and EXEC stays all-ones for WMMA).
// 256 threads = 8 waves (wave32): waves tiled 4(M) x 2(N), each 32x64 of C.
// ---------------------------------------------------------------------------
__global__ void __launch_bounds__(256)
wmma_gemm_kernel(const u16* __restrict__ Abase, long long lda, long long bsA,
                 const u16* __restrict__ Btbase, long long ldb, long long bsB,
                 const float* __restrict__ bias, long long bsBias,
                 const u16* __restrict__ extra, long long ldext, long long bsExt,
                 float scale, int do_elu,
                 u16* __restrict__ outb, float* __restrict__ outf,
                 long long ldc, long long bsC,
                 int M, int N, int K)
{
    // [buf0: A(10240B) B(10240B)][buf1: A B]
    __shared__ __align__(16) u16 sh[4 * TILE_ELems];

    const int z = blockIdx.z;
    const u16* A  = Abase  + (long long)z * bsA;
    const u16* Bt = Btbase + (long long)z * bsB;

    const int n0 = blockIdx.x * BN;
    const int m0 = blockIdx.y * BM;
    const int tid  = threadIdx.x;
    const int lane = tid & 31;
    const int wave = tid >> 5;
    const int wm = wave >> 1;   // 0..3 : 32-row slab
    const int wn = wave & 1;    // 0..1 : 64-col slab

    v8f acc[2][4];
#pragma unroll
    for (int mi = 0; mi < 2; ++mi)
#pragma unroll
        for (int ni = 0; ni < 4; ++ni)
#pragma unroll
            for (int j = 0; j < 8; ++j)
                acc[mi][ni][j] = 0.0f;

    // ---- async staging setup: 64 rows x 4 b128-chunks per pass, 2 passes ----
    const int lr = tid >> 2;          // 0..63
    const int lc = (tid & 3) * 8;     // bf16 col offset: 0,8,16,24

    const unsigned shbase = (unsigned)(unsigned long long)&sh[0];
    const unsigned offA0 = shbase + (unsigned)(lr * LDS_STRIDE + lc) * 2u;
    const unsigned offA1 = offA0 + 64u * LDS_STRIDE * 2u;
    const unsigned offB0 = offA0 + (unsigned)TILE_ELems * 2u;
    const unsigned offB1 = offB0 + 64u * LDS_STRIDE * 2u;

    const u16* gA0 = A + (long long)(m0 + lr) * lda + lc;
    const u16* gA1 = gA0 + 64ll * lda;
    int rb0 = n0 + lr;       if (rb0 > N - 1) rb0 = N - 1;   // clamp OOB cols (discarded later)
    int rb1 = n0 + lr + 64;  if (rb1 > N - 1) rb1 = N - 1;
    const u16* gB0 = Bt + (long long)rb0 * ldb + lc;
    const u16* gB1 = Bt + (long long)rb1 * ldb + lc;

    // fragment coordinates (ISA 7.12.2 16-bit layouts, wave32)
    const int rsel  = lane & 15;
    const int koffA = (lane < 16) ? 0 : 8;    // A: K {0..7,16..23} / {8..15,24..31}
    const int koffB = (lane < 16) ? 0 : 8;    // B: K {0..7,16..23} / {8..15,24..31}

    auto issue_stage = [&](int kk, unsigned bufsel) {
        const unsigned bo = bufsel * (unsigned)BUF_BYTES;
        async_b128(offA0 + bo, gA0 + kk);
        async_b128(offA1 + bo, gA1 + kk);
        async_b128(offB0 + bo, gB0 + kk);
        async_b128(offB1 + bo, gB1 + kk);
    };

    auto compute_stage = [&](int cur) {
        const u16* la = sh + (size_t)cur * (2 * TILE_ELems);
        const u16* lb = la + TILE_ELems;
        FragU a[2], b[4];
#pragma unroll
        for (int mi = 0; mi < 2; ++mi) {
            int row = wm * 32 + mi * 16 + rsel;
            a[mi].u[0] = *(const u32x4*)(la + row * LDS_STRIDE + koffA);
            a[mi].u[1] = *(const u32x4*)(la + row * LDS_STRIDE + koffA + 16);
        }
#pragma unroll
        for (int ni = 0; ni < 4; ++ni) {
            int row = wn * 64 + ni * 16 + rsel;
            b[ni].u[0] = *(const u32x4*)(lb + row * LDS_STRIDE + koffB);
            b[ni].u[1] = *(const u32x4*)(lb + row * LDS_STRIDE + koffB + 16);
        }
#pragma unroll
        for (int mi = 0; mi < 2; ++mi)
#pragma unroll
            for (int ni = 0; ni < 4; ++ni)
                acc[mi][ni] = __builtin_amdgcn_wmma_f32_16x16x32_bf16(
                    false, a[mi].v, false, b[ni].v, (short)0, acc[mi][ni],
                    false, false);
    };

    const int nk = K / BK;   // >= 2 for every GEMM in this network

    // prologue: stage 0 into buffer 0  (4 async instructions per wave)
    issue_stage(0, 0u);

    // steady state: branch-free issue/wait/compute
    for (int s = 0; s < nk - 1; ++s) {
        const int cur = s & 1;
        issue_stage((s + 1) * BK, (unsigned)(cur ^ 1));
        // 8 outstanding -> oldest 4 (stage s) complete when <= 4 remain
        asm volatile("s_wait_asynccnt 0x4" ::: "memory");
        __syncthreads();   // every wave signals only after ITS stage-s loads landed
        compute_stage(cur);
        __syncthreads();   // all reads of buffer `cur` done before it is re-filled
    }

    // peeled final stage: no further issues, no trailing barrier
    asm volatile("s_wait_asynccnt 0x0" ::: "memory");
    __syncthreads();
    compute_stage((nk - 1) & 1);

    // Epilogue. C/D layout: lanes 0-15 VGPR j -> (M=j, N=lane); lanes 16-31 -> (M=8+j, N=lane-16)
    const int mbase = m0 + wm * 32;
    const int nbase = n0 + wn * 64;
#pragma unroll
    for (int ni = 0; ni < 4; ++ni) {
        int n = nbase + ni * 16 + (lane & 15);
        if (n >= N) continue;
        float bv = bias ? bias[(long long)z * bsBias + n] : 0.0f;
#pragma unroll
        for (int mi = 0; mi < 2; ++mi) {
#pragma unroll
            for (int j = 0; j < 8; ++j) {
                int m = mbase + mi * 16 + ((lane < 16) ? j : 8 + j);
                float v = (acc[mi][ni][j] + bv) * scale;
                if (extra)
                    v += bf2f(extra[(long long)m * ldext + (long long)z * bsExt + n]);
                if (do_elu)
                    v = (v > 0.0f) ? v : (__expf(v) - 1.0f);
                long long idx = (long long)m * ldc + (long long)z * bsC + n;
                if (outb) outb[idx] = f2bf(v);
                if (outf) outf[idx] = v;
            }
        }
    }
}

// ---------------------------------------------------------------------------
// Host orchestration
// ---------------------------------------------------------------------------
extern "C" void kernel_launch(void* const* d_in, const int* in_sizes, int n_in,
                              void* d_out, int out_size, void* d_ws, size_t ws_size,
                              hipStream_t stream) {
    (void)in_sizes; (void)n_in; (void)out_size; (void)ws_size;

    constexpr int B = 1024, S = 512, H = 2048, R = 64, G = 16, L = 4, A_ = 256;
    constexpr float ADAPT = 0.1f;

    const float* s      = (const float*)d_in[0];
    const float* W_in   = (const float*)d_in[1];
    const float* b_in   = (const float*)d_in[2];
    const float* W_main = (const float*)d_in[3];
    const float* b_main = (const float*)d_in[4];
    const float* Wa     = (const float*)d_in[5];
    const float* ba     = (const float*)d_in[6];
    const float* Wb     = (const float*)d_in[7];
    const float* bb     = (const float*)d_in[8];
    const float* W_out  = (const float*)d_in[9];
    const float* b_out  = (const float*)d_in[10];
    float* out = (float*)d_out;

    // ---- carve scratch (bf16 buffers), 256B-aligned ----
    char* ws = (char*)d_ws;
    size_t off = 0;
    auto carve = [&](size_t bytes) -> u16* {
        u16* p = (u16*)(ws + off);
        off = (off + bytes + 255) & ~(size_t)255;
        return p;
    };
    u16* s_bf = carve((size_t)B * S * 2);
    u16* Wint = carve((size_t)H * S * 2);              // [H,S]
    u16* Wmt  = carve((size_t)L * H * H * 2);          // [L][H,H] (transposed)
    u16* Wat  = carve((size_t)L * G * R * H * 2);      // [L,G][R,H]
    u16* Wbt  = carve((size_t)L * G * H * R * 2);      // [L,G][H,R]
    u16* Wot  = carve((size_t)A_ * H * 2);             // [A,H]
    u16* X0   = carve((size_t)B * H * 2);              // group-invariant input activations
    u16* Xa   = carve((size_t)B * G * H * 2);          // ping
    u16* Xb   = carve((size_t)B * G * H * 2);          // pong
    u16* Hm   = carve((size_t)B * G * H * 2);          // main-branch pre-activation (bf16)
    u16* Z1   = carve((size_t)B * G * R * 2);          // LoRA down result

    // ---- weight conversion/transposition (~70 MB once per launch: trivial at 23 TB/s) ----
    {
        long long n = (long long)B * S;
        to_bf16_kernel<<<dim3((unsigned)((n + 255) / 256)), 256, 0, stream>>>(s, s_bf, n);
    }
    transpose_bf16_kernel<<<dim3(H / 32,  S / 32, 1),     256, 0, stream>>>(W_in,   Wint, S, H);
    transpose_bf16_kernel<<<dim3(H / 32,  H / 32, L),     256, 0, stream>>>(W_main, Wmt,  H, H);
    transpose_bf16_kernel<<<dim3(R / 32,  H / 32, L * G), 256, 0, stream>>>(Wa,     Wat,  H, R);
    transpose_bf16_kernel<<<dim3(H / 32,  R / 32, L * G), 256, 0, stream>>>(Wb,     Wbt,  R, H);
    transpose_bf16_kernel<<<dim3(A_ / 32, H / 32, 1),     256, 0, stream>>>(W_out,  Wot,  H, A_);

    auto gemm = [&](const u16* Ap, long long lda, long long bsA,
                    const u16* Btp, long long ldb, long long bsB,
                    const float* bias, long long bsBias,
                    const u16* extra, long long ldext, long long bsExt,
                    float scale, int elu,
                    u16* outb, float* outf, long long ldc, long long bsC,
                    int M, int N, int K, int batch) {
        dim3 grid((unsigned)((N + BN - 1) / BN), (unsigned)((M + BM - 1) / BM), (unsigned)batch);
        wmma_gemm_kernel<<<grid, 256, 0, stream>>>(
            Ap, lda, bsA, Btp, ldb, bsB, bias, bsBias,
            extra, ldext, bsExt, scale, elu,
            outb, outf, ldc, bsC, M, N, K);
    };

    // ---- input layer: X0[B,H] = s @ W_in + b_in (group-invariant, never broadcast) ----
    gemm(s_bf, S, 0, Wint, S, 0, b_in, 0,
         nullptr, 0, 0, 1.0f, 0,
         X0, nullptr, H, 0, B, H, S, 1);

    // ---- layer 0 (exploits group invariance: main GEMM is M=B, not M=B*G) ----
    gemm(X0, H, 0, Wmt, H, 0, b_main, 0,
         nullptr, 0, 0, 1.0f, 0,
         Hm, nullptr, H, 0, B, H, H, 1);
    gemm(X0, H, 0, Wat, H, (long long)R * H, ba, R,
         nullptr, 0, 0, 1.0f, 0,
         Z1, nullptr, (long long)G * R, R, B, R, H, G);
    gemm(Z1, (long long)G * R, R, Wbt, R, (long long)H * R, bb, H,
         Hm, H, 0, ADAPT, 1,
         Xa, nullptr, (long long)G * H, H, B, H, R, G);

    u16* Xc = Xa;
    u16* Xn = Xb;
    for (int i = 1; i < L; ++i) {
        gemm(Xc, H, 0, Wmt + (size_t)i * H * H, H, 0, b_main + (size_t)i * H, 0,
             nullptr, 0, 0, 1.0f, 0,
             Hm, nullptr, H, 0, B * G, H, H, 1);
        gemm(Xc, (long long)G * H, H, Wat + (size_t)i * G * R * H, H, (long long)R * H,
             ba + (size_t)i * G * R, R,
             nullptr, 0, 0, 1.0f, 0,
             Z1, nullptr, (long long)G * R, R, B, R, H, G);
        gemm(Z1, (long long)G * R, R, Wbt + (size_t)i * G * H * R, R, (long long)H * R,
             bb + (size_t)i * G * H, H,
             Hm, (long long)G * H, H, ADAPT, 1,
             Xn, nullptr, (long long)G * H, H, B, H, R, G);
        u16* t = Xc; Xc = Xn; Xn = t;
    }

    // ---- logits[BG,A] = X @ W_out + b_out (f32 to d_out) ----
    gemm(Xc, H, 0, Wot, H, 0, b_out, 0,
         nullptr, 0, 0, 1.0f, 0,
         nullptr, out, A_, 0, B * G, A_, H, 1);
}
